// CNN_24472723653055
// MI455X (gfx1250) — matrix operand order repaired and checked by reading, not compiled, not run on hardware
//
#include <hip/hip_runtime.h>
#include <hip/hip_bf16.h>
#include <hip/hip_fp16.h>

typedef __attribute__((ext_vector_type(16))) _Float16 v16h;
typedef __attribute__((ext_vector_type(8)))  float    v8f;

#define EPS_BN 1e-5f
#define EPS_LN 1e-5f
#define EPS_L2 1e-12f

// ---------------- Layer 1: adder conv (Cin=1, kh=6, kw=1, stride 2x1) ----------------
// x: (16,1,196,3)  w: (128,1,6,1)  out: (16,128,96,3) NCHW
__global__ void adder_l1(const float* __restrict__ x, const float* __restrict__ w,
                         float* __restrict__ out) {
  int idx = blockIdx.x * blockDim.x + threadIdx.x;
  const int total = 16 * 128 * 96 * 3;
  if (idx >= total) return;
  int wo = idx % 3;  int t = idx / 3;
  int ho = t % 96;   t /= 96;
  int co = t % 128;  int n = t / 128;
  const float* xp = x + n * (196 * 3) + wo;
  const float* wp = w + co * 6;
  float s = 0.f;
#pragma unroll
  for (int i = 0; i < 6; ++i)
    s += fabsf(xp[(2 * ho + i) * 3] - wp[i]);
  out[idx] = -s;
}

// ---------------- Layer 2: adder conv (Cin=128, kh=6, kw=1, stride 2x1) ----------------
// in: (16,128,96,3)  w: (256,128,6,1)  out: (16,256,46,3)
// One block per (n,ho,wo); patch (128*6 floats) staged in LDS; one thread per Cout.
__global__ void adder_l2(const float* __restrict__ in, const float* __restrict__ w,
                         float* __restrict__ out) {
  __shared__ float patch[128 * 6];
  int b = blockIdx.x;
  int wo = b % 3;  b /= 3;
  int ho = b % 46; int n = b / 46;
  for (int k = threadIdx.x; k < 128 * 6; k += blockDim.x) {
    int ci = k / 6, i = k - ci * 6;
    patch[k] = in[((n * 128 + ci) * 96 + (2 * ho + i)) * 3 + wo];
  }
  __syncthreads();
  int co = threadIdx.x;            // blockDim.x == 256
  const float* wrow = w + co * (128 * 6);
  __builtin_prefetch(wrow, 0, 1);  // global_prefetch_b8
  float s = 0.f;
#pragma unroll 6
  for (int k = 0; k < 128 * 6; ++k)
    s += fabsf(patch[k] - wrow[k]);
  out[((n * 256 + co) * 46 + ho) * 3 + wo] = -s;
}

// ---------------- Layer 3: adder conv (Cin=256, kh=6, kw=2, stride 2x1) ----------------
// in: (16,256,46,3)  w: (384,256,6,2)  out: (16,384,21,2)
__global__ void adder_l3(const float* __restrict__ in, const float* __restrict__ w,
                         float* __restrict__ out) {
  __shared__ float patch[256 * 12];      // ci*12 + i*2 + j, matching w layout
  int b = blockIdx.x;
  int wo = b % 2;  b /= 2;
  int ho = b % 21; int n = b / 21;
  for (int k = threadIdx.x; k < 256 * 12; k += blockDim.x) {
    int ci = k / 12; int r = k - ci * 12; int i = r >> 1; int j = r & 1;
    patch[k] = in[((n * 256 + ci) * 46 + (2 * ho + i)) * 3 + (wo + j)];
  }
  __syncthreads();
  int co = threadIdx.x;            // blockDim.x == 384
  const float* wrow = w + co * (256 * 12);
  __builtin_prefetch(wrow, 0, 1);
  float s = 0.f;
#pragma unroll 4
  for (int k = 0; k < 256 * 12; ++k)
    s += fabsf(patch[k] - wrow[k]);
  out[((n * 384 + co) * 21 + ho) * 2 + wo] = -s;
}

// ---------------- BatchNorm stats: one block per channel ----------------
__global__ void bn_stats(const float* __restrict__ data, int C, int HW,
                         float* __restrict__ mean, float* __restrict__ rstd) {
  __shared__ float ssum[256];
  __shared__ float ssq[256];
  int c = blockIdx.x;
  int per = 16 * HW;
  float s = 0.f, q = 0.f;
  for (int e = threadIdx.x; e < per; e += blockDim.x) {
    int n = e / HW, p = e - n * HW;
    float v = data[(n * C + c) * HW + p];
    s += v; q += v * v;
  }
  ssum[threadIdx.x] = s; ssq[threadIdx.x] = q;
  __syncthreads();
  for (int off = 128; off > 0; off >>= 1) {
    if (threadIdx.x < off) {
      ssum[threadIdx.x] += ssum[threadIdx.x + off];
      ssq[threadIdx.x]  += ssq[threadIdx.x + off];
    }
    __syncthreads();
  }
  if (threadIdx.x == 0) {
    float mu  = ssum[0] / (float)per;
    float var = ssq[0] / (float)per - mu * mu;
    mean[c] = mu;
    rstd[c] = rsqrtf(var + EPS_BN);
  }
}

// ---------------- BN + ReLU apply (in place, f32) ----------------
__global__ void bn_apply(float* __restrict__ data,
                         const float* __restrict__ mean, const float* __restrict__ rstd,
                         const float* __restrict__ gamma, const float* __restrict__ beta,
                         int C, int HW, int total) {
  int idx = blockIdx.x * blockDim.x + threadIdx.x;
  if (idx >= total) return;
  int c = (idx / HW) % C;
  float v = data[idx];
  v = (v - mean[c]) * rstd[c] * gamma[c] + beta[c];
  data[idx] = fmaxf(v, 0.f);
}

// ---------------- BN + ReLU apply with f16 output (feeds WMMA FC) ----------------
__global__ void bn_apply_f16(const float* __restrict__ data,
                             const float* __restrict__ mean, const float* __restrict__ rstd,
                             const float* __restrict__ gamma, const float* __restrict__ beta,
                             int C, int HW, int total, _Float16* __restrict__ out) {
  int idx = blockIdx.x * blockDim.x + threadIdx.x;
  if (idx >= total) return;
  int c = (idx / HW) % C;
  float v = data[idx];
  v = (v - mean[c]) * rstd[c] * gamma[c] + beta[c];
  out[idx] = (_Float16)fmaxf(v, 0.f);
}

// ---------------- Pad/convert Wfc (6,16128) f32 -> Wfcp (16,16128) f16, rows 6..15 = 0 ----
__global__ void pad_wfc(const float* __restrict__ Wfc, _Float16* __restrict__ Wfcp) {
  int idx = blockIdx.x * blockDim.x + threadIdx.x;
  const int total = 16 * 16128;
  if (idx >= total) return;
  int row = idx / 16128;
  int col = idx - row * 16128;
  Wfcp[idx] = (row < 6) ? (_Float16)Wfc[row * 16128 + col] : (_Float16)0.f;
}

// ---------------- FC head via WMMA + fused LayerNorm + L2 normalize ----------------
// h3:   (16, 16128) f16 (NCHW flatten).
// Wfcp: (16, 16128) f16, rows >=6 zero (B is K x N with N = row index here).
// out:  (16,6) = 96 floats.
// Single wave32: M=16 rows = batch, N cols padded 6->16, K=16128 in chunks of 32.
// Inner loop: 4x global_load_b128 (A halves + B halves) + 1x v_wmma.
__global__ void fc_head(const _Float16* __restrict__ h3, const _Float16* __restrict__ Wfcp,
                        const float* __restrict__ bfc, float* __restrict__ out) {
  __shared__ float hmat[96];
  __shared__ float hnorm[96];
  int lane = threadIdx.x;        // 0..31, full wave, EXEC all-1s at WMMA
  int g    = lane >> 4;          // lane-group select
  int mn   = lane & 15;          // A-row M / B-col N / D-col N
  const _Float16* arow = h3   + mn * 16128 + 8 * g;    // halfs 0..7 at +0, 8..15 at +16
  const _Float16* brow = Wfcp + mn * 16128 + 8 * g;    // halfs 0..7 at +0, 8..15 at +16
  v8f acc = {};
  for (int kb = 0; kb < 16128; kb += 32) {
    v16h a, b;
    // 16-bit A layout (16x32): halfs i -> K = 8*g + (i&7) + 16*(i>=8)
#pragma unroll
    for (int i = 0; i < 8; ++i) {
      a[i]     = arow[kb + i];
      a[i + 8] = arow[kb + 16 + i];
    }
    // 16-bit B layout (32x16): halfs i -> K = 8*g + (i&7) + 16*(i>=8)
#pragma unroll
    for (int i = 0; i < 8; ++i) {
      b[i]     = brow[kb + i];
      b[i + 8] = brow[kb + 16 + i];
    }
    acc = __builtin_amdgcn_wmma_f32_16x16x32_f16(
        /*neg_a=*/false, a, /*neg_b=*/false, b,
        /*c_mod=*/(short)0, acc, /*reuse_a=*/false, /*reuse_b=*/false);
  }
  // D layout: VGPR r holds row M = r + 8*g, col N = lane&15
  if (mn < 6) {
#pragma unroll
    for (int r = 0; r < 8; ++r)
      hmat[(r + 8 * g) * 6 + mn] = acc[r] + bfc[mn];
  }
  __syncthreads();
  // Global LayerNorm over all 96 entries (biased variance), every lane redundantly
  float s = 0.f, q = 0.f;
  for (int i = 0; i < 96; ++i) { float v = hmat[i]; s += v; q += v * v; }
  float mu  = s * (1.f / 96.f);
  float var = q * (1.f / 96.f) - mu * mu;
  float rs  = rsqrtf(var + EPS_LN);
  for (int i = lane; i < 96; i += 32) hnorm[i] = (hmat[i] - mu) * rs;
  __syncthreads();
  // Per-row L2 normalize
  for (int i = lane; i < 96; i += 32) {
    int row = i / 6;
    float nn = 0.f;
#pragma unroll
    for (int j = 0; j < 6; ++j) { float v = hnorm[row * 6 + j]; nn += v * v; }
    out[i] = hnorm[i] / fmaxf(sqrtf(nn), EPS_L2);
  }
}

extern "C" void kernel_launch(void* const* d_in, const int* in_sizes, int n_in,
                              void* d_out, int out_size, void* d_ws, size_t ws_size,
                              hipStream_t stream) {
  (void)in_sizes; (void)n_in; (void)out_size; (void)ws_size;
  const float* x   = (const float*)d_in[0];
  const float* w1  = (const float*)d_in[1];
  const float* g1  = (const float*)d_in[2];
  const float* b1  = (const float*)d_in[3];
  const float* w2  = (const float*)d_in[4];
  const float* g2  = (const float*)d_in[5];
  const float* b2  = (const float*)d_in[6];
  const float* w3  = (const float*)d_in[7];
  const float* g3  = (const float*)d_in[8];
  const float* b3  = (const float*)d_in[9];
  const float* Wfc = (const float*)d_in[10];
  const float* bfc = (const float*)d_in[11];
  float* out = (float*)d_out;

  const int N1 = 16 * 128 * 96 * 3;   // 589824
  const int N2 = 16 * 256 * 46 * 3;   // 565248
  const int N3 = 16 * 384 * 21 * 2;   // 258048
  const int NW = 16 * 16128;          // 258048 (padded Wfc)

  char* ws = (char*)d_ws;
  size_t off = 0;
  auto carve = [&](size_t bytes) -> char* {
    char* p = ws + off;
    off += (bytes + 255) & ~(size_t)255;
    return p;
  };
  float*    h1    = (float*)   carve((size_t)N1 * 4);
  float*    h2    = (float*)   carve((size_t)N2 * 4);
  float*    h3    = (float*)   carve((size_t)N3 * 4);
  _Float16* h3h   = (_Float16*)carve((size_t)N3 * 2);
  _Float16* Wfcp  = (_Float16*)carve((size_t)NW * 2);
  float*    mean1 = (float*)   carve(128 * 4);
  float*    rstd1 = (float*)   carve(128 * 4);
  float*    mean2 = (float*)   carve(256 * 4);
  float*    rstd2 = (float*)   carve(256 * 4);
  float*    mean3 = (float*)   carve(384 * 4);
  float*    rstd3 = (float*)   carve(384 * 4);

  // Padded f16 FC weights (independent of the conv pipeline)
  pad_wfc<<<(NW + 255) / 256, 256, 0, stream>>>(Wfc, Wfcp);
  // Layer 1
  adder_l1<<<(N1 + 255) / 256, 256, 0, stream>>>(x, w1, h1);
  bn_stats<<<128, 256, 0, stream>>>(h1, 128, 96 * 3, mean1, rstd1);
  bn_apply<<<(N1 + 255) / 256, 256, 0, stream>>>(h1, mean1, rstd1, g1, b1, 128, 96 * 3, N1);
  // Layer 2
  adder_l2<<<16 * 46 * 3, 256, 0, stream>>>(h1, w2, h2);
  bn_stats<<<256, 256, 0, stream>>>(h2, 256, 46 * 3, mean2, rstd2);
  bn_apply<<<(N2 + 255) / 256, 256, 0, stream>>>(h2, mean2, rstd2, g2, b2, 256, 46 * 3, N2);
  // Layer 3
  adder_l3<<<16 * 21 * 2, 384, 0, stream>>>(h2, w3, h3);
  bn_stats<<<384, 256, 0, stream>>>(h3, 384, 21 * 2, mean3, rstd3);
  bn_apply_f16<<<(N3 + 255) / 256, 256, 0, stream>>>(h3, mean3, rstd3, g3, b3, 384, 21 * 2, N3, h3h);
  // FC + LayerNorm + L2 (single-wave WMMA kernel)
  fc_head<<<1, 32, 0, stream>>>(h3h, Wfcp, bfc, out);
}
